// DeformableAttn_3410204033225
// MI455X (gfx1250) — matrix operand order repaired and checked by reading, not compiled, not run on hardware
//
#include <hip/hip_runtime.h>

// ---------------- problem constants ----------------
#define FD    256
#define NHD   4
#define EDIM  64
#define SP3N  8
#define NP    32          // SP3N * NHD aux points per query
#define HRES  128
#define BSZ   4
#define NSQ   1024
#define BN    4096        // BSZ * NSQ

// packed B-fragment table: frag = base + kstep*ntiles + ntile, 512 halves each
#define PK_OFF_BASE 0     // w_off: 8 ksteps x 6 ntiles (ld=96)
#define PK_Q_BASE   48    // w_q:   8 x 4 (ld=64)
#define PK_K_BASE   80    // w_k:   8 x 4
#define PK_V_BASE   112   // w_v:   8 x 4
#define PK_OUT_BASE 144   // w_out: 8 x 16 (ld=256)
#define PK_TOTAL    272

typedef __attribute__((ext_vector_type(16))) _Float16 v16h;
typedef __attribute__((ext_vector_type(8)))  _Float16 v8h;
typedef __attribute__((ext_vector_type(8)))  float    v8f;

__device__ __forceinline__ int imin(int a, int b) { return a < b ? a : b; }

// ---------------- WMMA fragment helpers (wave32, 16x16x32 f16) ----------------
// A (16x32 f16), row-major source with leading dim ld:
//   lanes 0-15: M=lane, VGPR0-3 K=0..7, VGPR4-7 K=16..23
//   lanes16-31: M=lane-16, VGPR0-3 K=8..15, VGPR4-7 K=24..31
__device__ __forceinline__ v16h load_frag_a(const _Float16* src, int ld) {
  int lane = threadIdx.x & 31;
  int m  = lane & 15;
  int k0 = (lane < 16) ? 0 : 8;
  const _Float16* r = src + m * ld + k0;
  v8h lo = *(const v8h*)(r);        // K = k0..k0+7      (16B aligned)
  v8h hi = *(const v8h*)(r + 16);   // K = k0+16..k0+23
  return __builtin_shufflevector(lo, hi, 0, 1, 2, 3, 4, 5, 6, 7,
                                 8, 9, 10, 11, 12, 13, 14, 15);
}

// B fragment from the pre-packed register-layout table: one 32B load per lane
__device__ __forceinline__ v16h load_frag_b_pk(const _Float16* pk, int frag) {
  int lane = threadIdx.x & 31;
  return *(const v16h*)(pk + (size_t)frag * 512 + lane * 16);
}

#define WMMA_F16(a, b, c) \
  __builtin_amdgcn_wmma_f32_16x16x32_f16(false, (a), false, (b), (short)0, (c), false, false)

// ---------------- bilinear setup (align_corners=True, clip to [0,1]) ----------------
__device__ __forceinline__ void bilin_setup(float u, float v,
                                            int& i00, int& i01, int& i10, int& i11,
                                            float& w00, float& w01, float& w10, float& w11) {
  u = fminf(fmaxf(u, 0.f), 1.f) * (float)(HRES - 1);
  v = fminf(fmaxf(v, 0.f), 1.f) * (float)(HRES - 1);
  float xf = floorf(u), yf = floorf(v);
  int x0 = (int)xf, y0 = (int)yf;
  int x1 = imin(x0 + 1, HRES - 1);
  int y1 = imin(y0 + 1, HRES - 1);
  float wx = u - xf, wy = v - yf;
  i00 = (y0 * HRES + x0) * FD;  i01 = (y0 * HRES + x1) * FD;
  i10 = (y1 * HRES + x0) * FD;  i11 = (y1 * HRES + x1) * FD;
  w00 = (1.f - wy) * (1.f - wx); w01 = (1.f - wy) * wx;
  w10 = wy * (1.f - wx);         w11 = wy * wx;
}

// single-channel tap from channel-last f16 plane
__device__ __forceinline__ float sample_one(const _Float16* base, float u, float v, int c) {
  int i00, i01, i10, i11; float w00, w01, w10, w11;
  bilin_setup(u, v, i00, i01, i10, i11, w00, w01, w10, w11);
  return w00 * (float)base[i00 + c] + w01 * (float)base[i01 + c]
       + w10 * (float)base[i10 + c] + w11 * (float)base[i11 + c];
}

// 8-channel vectorized tap (lane handles channels lane*8 .. lane*8+7)
__device__ __forceinline__ void sample_accum8(const _Float16* base, float u, float v,
                                              int lane, float* acc) {
  int i00, i01, i10, i11; float w00, w01, w10, w11;
  bilin_setup(u, v, i00, i01, i10, i11, w00, w01, w10, w11);
  const _Float16* bp = base + lane * 8;
  v8h c00 = *(const v8h*)(bp + i00);
  v8h c01 = *(const v8h*)(bp + i01);
  v8h c10 = *(const v8h*)(bp + i10);
  v8h c11 = *(const v8h*)(bp + i11);
#pragma unroll
  for (int i = 0; i < 8; ++i)
    acc[i] += w00 * (float)c00[i] + w01 * (float)c01[i]
            + w10 * (float)c10[i] + w11 * (float)c11[i];
}

// ================= K0a: pack all weights into WMMA B-fragment layout (f16) ========
__global__ void __launch_bounds__(256) pack_weights(
    const float* __restrict__ w_off, const float* __restrict__ w_q,
    const float* __restrict__ w_k,  const float* __restrict__ w_v,
    const float* __restrict__ w_out, _Float16* __restrict__ pk) {
  int frag = blockIdx.x * 8 + (threadIdx.x >> 5);      // 272 fragments, 34 blocks
  int lane = threadIdx.x & 31;
  if (frag >= PK_TOTAL) return;
  const float* w; int ld; int f = frag;
  if (f < PK_Q_BASE)        { w = w_off; ld = 96; }
  else if (f < PK_K_BASE)   { w = w_q;   ld = 64;  f -= PK_Q_BASE; }
  else if (f < PK_V_BASE)   { w = w_k;   ld = 64;  f -= PK_K_BASE; }
  else if (f < PK_OUT_BASE) { w = w_v;   ld = 64;  f -= PK_V_BASE; }
  else                      { w = w_out; ld = 256; f -= PK_OUT_BASE; }
  int ntiles = ld / 16;
  int ks = f / ntiles, nt = f % ntiles;
  int n  = lane & 15;
  int kb = (lane < 16) ? 0 : 8;
  const float* src = w + (size_t)(ks * 32 + kb) * ld + nt * 16 + n;
  v16h b;
#pragma unroll
  for (int i = 0; i < 16; ++i) b[i] = (_Float16)src[(size_t)(i < 8 ? i : i + 8) * ld];
  *(v16h*)(pk + (size_t)frag * 512 + lane * 16) = b;   // 32B/lane, coalesced 1KB/wave
}

// ================= K0b: plane transpose [B,C,H,W]f32 -> [B,H,W,C]f16 =================
__global__ void __launch_bounds__(256) transpose_planes(
    const float* __restrict__ p0, const float* __restrict__ p1, const float* __restrict__ p2,
    _Float16* __restrict__ o0, _Float16* __restrict__ o1, _Float16* __restrict__ o2) {
  __shared__ __align__(16) _Float16 tile[32 * 264];   // padded vs bank conflicts
  int id = blockIdx.x;
  int xs = id & 3;  id >>= 2;           // 4 strips of 32 in x
  int y  = id & 127; id >>= 7;
  int b  = id & 3;  id >>= 2;
  int pl = id;                          // 0..2
  const float* in  = (pl == 0) ? p0 : ((pl == 1) ? p1 : p2);
  _Float16*    out = (pl == 0) ? o0 : ((pl == 1) ? o1 : o2);
  int l  = threadIdx.x & 31;            // x offset within strip (coalesced read)
  int wv = threadIdx.x >> 5;            // 0..7 channel group
  const float* src = in + (((size_t)b * FD) * HRES + y) * HRES + xs * 32 + l;
#pragma unroll
  for (int cg = 0; cg < 32; ++cg) {
    int c = cg * 8 + wv;
    tile[l * 264 + c] = (_Float16)src[(size_t)c * HRES * HRES];
  }
  __syncthreads();
  int t = threadIdx.x;                  // channel (coalesced write)
  _Float16* dst = out + (((size_t)b * HRES + y) * HRES + xs * 32) * FD + t;
#pragma unroll
  for (int x = 0; x < 32; ++x) dst[x * FD] = tile[x * 264 + t];
}

// ================= K1: base triplane features =================
__global__ void __launch_bounds__(256) base_feature(
    const float* __restrict__ qp,
    const _Float16* __restrict__ p0, const _Float16* __restrict__ p1,
    const _Float16* __restrict__ p2,
    float* __restrict__ feat32, _Float16* __restrict__ feat16) {
  int q = blockIdx.x;        // 0..4095
  int c = threadIdx.x;       // channel
  int b = q >> 10;
  float px = qp[q * 3 + 0], py = qp[q * 3 + 1], pz = qp[q * 3 + 2];
  size_t boff = (size_t)b * HRES * HRES * FD;
  float acc = sample_one(p0 + boff, px, pz, c)     // c_xz: (x=px, y=pz)
            + sample_one(p1 + boff, px, py, c)     // c_xy: (x=px, y=py)
            + sample_one(p2 + boff, py, pz, c);    // c_yz: (x=py, y=pz)
  feat32[(size_t)q * FD + c] = acc;
  feat16[(size_t)q * FD + c] = (_Float16)acc;
}

// ================= K2: offsets + Q projection (WMMA), builds aux positions =================
__global__ void __launch_bounds__(256) proj_off_q(
    const _Float16* __restrict__ feat16, const float* __restrict__ qp,
    const _Float16* __restrict__ pk,
    const float* __restrict__ b_off, const float* __restrict__ b_q,
    float* __restrict__ auxp, float* __restrict__ qvec) {
  int wv   = blockIdx.x * 8 + (threadIdx.x >> 5);  // 2560 waves = 256 Mtiles x 10 Ntiles
  int lane = threadIdx.x & 31;
  int mt = wv / 10, nt = wv % 10;                  // nt 0..5 -> offsets, 6..9 -> q
  v8f acc = {};
#pragma unroll
  for (int ks = 0; ks < 8; ++ks) {
    v16h a = load_frag_a(feat16 + (size_t)(mt * 16) * FD + ks * 32, FD);
    int frag = (nt < 6) ? (PK_OFF_BASE + ks * 6 + nt)
                        : (PK_Q_BASE + ks * 4 + (nt - 6));
    v16h bf = load_frag_b_pk(pk, frag);
    acc = WMMA_F16(a, bf, acc);
  }
  int n = lane & 15;
#pragma unroll
  for (int r = 0; r < 8; ++r) {
    int m   = (lane < 16) ? r : r + 8;
    int row = mt * 16 + m;
    if (nt < 6) {
      int col = nt * 16 + n;                // 0..95 = [j][h][d]
      int d   = col % 3;
      auxp[(size_t)row * 96 + col] = acc[r] + b_off[col] + qp[row * 3 + d];
    } else {
      int col = (nt - 6) * 16 + n;          // 0..63
      qvec[(size_t)row * EDIM + col] = (acc[r] + b_q[col]) * 8.0f;  // * sqrt(E)
    }
  }
}

// ================= K3: fused aux-sampling + K/V WMMA + attention =================
__global__ void __launch_bounds__(256, 1) deform_attn(
    const _Float16* __restrict__ p0, const _Float16* __restrict__ p1,
    const _Float16* __restrict__ p2,
    const float* __restrict__ auxp, const float* __restrict__ qvec,
    const _Float16* __restrict__ pk,
    const float* __restrict__ b_k, const float* __restrict__ b_v,
    _Float16* __restrict__ attn16) {
  __shared__ __align__(16) _Float16 af[NP * FD];   // 32x256 f16 aux features (16 KB)
  __shared__ float k_lds[NP * EDIM];               // 8 KB
  __shared__ float v_lds[NP * EDIM];               // 8 KB
  __shared__ float sim[NP];
  __shared__ float att[NP];
  int q    = blockIdx.x;
  int b    = q >> 10;
  int lane = threadIdx.x & 31;
  int wv   = threadIdx.x >> 5;
  size_t boff = (size_t)b * HRES * HRES * FD;
  const _Float16* pb0 = p0 + boff;
  const _Float16* pb1 = p1 + boff;
  const _Float16* pb2 = p2 + boff;
  // warm L2 for the packed K/V weight fragments
  __builtin_prefetch(pk + (size_t)PK_K_BASE * 512 + (size_t)threadIdx.x * 128, 0, 1);

  // ---- sampling: wave handles points p = wv + 8*pp, lane handles 8 channels ----
#pragma unroll
  for (int pp = 0; pp < 4; ++pp) {
    int p = wv + 8 * pp;
    float px = auxp[(size_t)q * 96 + p * 3 + 0];
    float py = auxp[(size_t)q * 96 + p * 3 + 1];
    float pz = auxp[(size_t)q * 96 + p * 3 + 2];
    float acc[8] = {0.f, 0.f, 0.f, 0.f, 0.f, 0.f, 0.f, 0.f};
    sample_accum8(pb0, px, pz, lane, acc);
    sample_accum8(pb1, px, py, lane, acc);
    sample_accum8(pb2, py, pz, lane, acc);
#pragma unroll
    for (int i = 0; i < 8; ++i) af[p * FD + lane * 8 + i] = (_Float16)acc[i];
  }
  __syncthreads();

  // ---- K and V projections: 8 wave-tiles (M=32 -> 2, N=64 -> 4) ----
  {
    int mt = wv >> 2, nt = wv & 3;
    v8f ak = {}, av = {};
#pragma unroll
    for (int ks = 0; ks < 8; ++ks) {
      v16h a  = load_frag_a(af + mt * 16 * FD + ks * 32, FD);
      v16h bk = load_frag_b_pk(pk, PK_K_BASE + ks * 4 + nt);
      ak = WMMA_F16(a, bk, ak);
      v16h bv = load_frag_b_pk(pk, PK_V_BASE + ks * 4 + nt);
      av = WMMA_F16(a, bv, av);
    }
    int n = lane & 15;
#pragma unroll
    for (int r = 0; r < 8; ++r) {
      int m = (lane < 16) ? r : r + 8;
      int row = mt * 16 + m, col = nt * 16 + n;
      k_lds[row * EDIM + col] = ak[r] + b_k[col];
      v_lds[row * EDIM + col] = av[r] + b_v[col];
    }
  }
  __syncthreads();

  // ---- sim = q . k  (p = j*NH + h) ----
  int t = threadIdx.x;
  if (t < NP) {
    float s = 0.f;
#pragma unroll 8
    for (int e = 0; e < EDIM; ++e) s += qvec[(size_t)q * EDIM + e] * k_lds[t * EDIM + e];
    sim[t] = s;
  }
  __syncthreads();
  // ---- softmax over j per head ----
  if (t < NHD) {
    float mx = -1e30f;
#pragma unroll
    for (int j = 0; j < SP3N; ++j) mx = fmaxf(mx, sim[j * NHD + t]);
    float ssum = 0.f, ex[SP3N];
#pragma unroll
    for (int j = 0; j < SP3N; ++j) { ex[j] = __expf(sim[j * NHD + t] - mx); ssum += ex[j]; }
    float inv = 1.0f / ssum;
#pragma unroll
    for (int j = 0; j < SP3N; ++j) att[j * NHD + t] = ex[j] * inv;
  }
  __syncthreads();
  // ---- out[h][e] = sum_j attn * v ----
  {
    int h = t >> 6, e = t & 63;
    float o = 0.f;
#pragma unroll
    for (int j = 0; j < SP3N; ++j) o += att[j * NHD + h] * v_lds[(j * NHD + h) * EDIM + e];
    attn16[(size_t)q * FD + h * EDIM + e] = (_Float16)o;
  }
}

// ================= K4: output projection + bias + residual =================
__global__ void __launch_bounds__(256) out_proj(
    const _Float16* __restrict__ attn16, const _Float16* __restrict__ pk,
    const float* __restrict__ b_out, const float* __restrict__ feat32,
    float* __restrict__ out) {
  int wv   = blockIdx.x * 8 + (threadIdx.x >> 5);   // 4096 waves = 256 x 16 tiles
  int lane = threadIdx.x & 31;
  int mt = wv >> 4, nt = wv & 15;
  v8f acc = {};
#pragma unroll
  for (int ks = 0; ks < 8; ++ks) {
    v16h a = load_frag_a(attn16 + (size_t)(mt * 16) * FD + ks * 32, FD);
    v16h b = load_frag_b_pk(pk, PK_OUT_BASE + ks * 16 + nt);
    acc = WMMA_F16(a, b, acc);
  }
  int n = lane & 15;
#pragma unroll
  for (int r = 0; r < 8; ++r) {
    int m = (lane < 16) ? r : r + 8;
    int row = mt * 16 + m, col = nt * 16 + n;
    out[(size_t)row * FD + col] = acc[r] + b_out[col] + feat32[(size_t)row * FD + col];
  }
}

// ================= host: wire up workspace and launch =================
extern "C" void kernel_launch(void* const* d_in, const int* in_sizes, int n_in,
                              void* d_out, int out_size, void* d_ws, size_t ws_size,
                              hipStream_t stream) {
  (void)in_sizes; (void)n_in; (void)out_size; (void)ws_size;
  const float* qp    = (const float*)d_in[0];
  const float* c_xz  = (const float*)d_in[1];
  const float* c_xy  = (const float*)d_in[2];
  const float* c_yz  = (const float*)d_in[3];
  const float* w_off = (const float*)d_in[4];
  const float* b_off = (const float*)d_in[5];
  const float* w_q   = (const float*)d_in[6];
  const float* b_q   = (const float*)d_in[7];
  const float* w_k   = (const float*)d_in[8];
  const float* b_k   = (const float*)d_in[9];
  const float* w_v   = (const float*)d_in[10];
  const float* b_v   = (const float*)d_in[11];
  const float* w_out = (const float*)d_in[12];
  const float* b_out = (const float*)d_in[13];
  float* out = (float*)d_out;

  // workspace partition (~112 MB total)
  const size_t plane_elems = (size_t)BSZ * HRES * HRES * FD;   // 16.7M halves/plane
  _Float16* pt0 = (_Float16*)d_ws;
  _Float16* pt1 = pt0 + plane_elems;
  _Float16* pt2 = pt1 + plane_elems;
  float*    feat32 = (float*)(pt2 + plane_elems);
  _Float16* feat16 = (_Float16*)(feat32 + (size_t)BN * FD);
  float*    qvec   = (float*)(feat16 + (size_t)BN * FD);
  float*    auxp   = qvec + (size_t)BN * EDIM;
  _Float16* attn16 = (_Float16*)(auxp + (size_t)BN * NP * 3);
  _Float16* pkw    = attn16 + (size_t)BN * FD;                 // 272 frags * 512 halves

  pack_weights<<<34, 256, 0, stream>>>(w_off, w_q, w_k, w_v, w_out, pkw);
  transpose_planes<<<3 * BSZ * HRES * 4, 256, 0, stream>>>(c_xz, c_xy, c_yz, pt0, pt1, pt2);
  base_feature<<<BN, 256, 0, stream>>>(qp, pt0, pt1, pt2, feat32, feat16);
  proj_off_q<<<320, 256, 0, stream>>>(feat16, qp, pkw, b_off, b_q, auxp, qvec);
  deform_attn<<<BN, 256, 0, stream>>>(pt0, pt1, pt2, auxp, qvec, pkw, b_k, b_v, attn16);
  out_proj<<<512, 256, 0, stream>>>(attn16, pkw, b_out, feat32, out);
}
